// Net_27891517620298
// MI455X (gfx1250) — matrix operand order repaired and checked by reading, not compiled, not run on hardware
//
#include <hip/hip_runtime.h>

typedef unsigned short u16;
typedef unsigned int   u32;

typedef __attribute__((ext_vector_type(16))) __bf16 v16bf;
typedef __attribute__((ext_vector_type(8)))  float  v8f;

// ---------------- bf16 helpers (round-to-nearest-even) ----------------
__device__ __forceinline__ u16 f2bf(float f) {
  u32 u = __float_as_uint(f);
  u32 r = u + 0x7FFFu + ((u >> 16) & 1u);
  return (u16)(r >> 16);
}
__device__ __forceinline__ float bf2f(u16 h) {
  return __uint_as_float(((u32)h) << 16);
}

template<typename T> __device__ __forceinline__ u16 toBF(T v);
template<> __device__ __forceinline__ u16 toBF<float>(float v) { return f2bf(v); }
template<> __device__ __forceinline__ u16 toBF<u16>(u16 v)     { return v; }

template<typename T> __device__ __forceinline__ T fromF(float v);
template<> __device__ __forceinline__ u16   fromF<u16>(float v)   { return f2bf(v); }
template<> __device__ __forceinline__ float fromF<float>(float v) { return v; }

// ---------------- CDNA5 async LDS staging (ISA 08_async_tensor) ----------------
// 16B global -> 16B LDS, tracked by ASYNCcnt. LDS address = low 32 bits of the
// flat address (ISA 10.2).
__device__ __forceinline__ void async_ld16(void* ldsDst, const void* gsrc) {
  u32 l = (u32)(size_t)ldsDst;
  asm volatile("global_load_async_to_lds_b128 %0, %1, off"
               :: "v"(l), "v"(gsrc) : "memory");
}
__device__ __forceinline__ void wait_asynccnt0() {
  asm volatile("s_wait_asynccnt 0" ::: "memory");
}

// ---------------- mask pipeline: histogram radix-select ----------------
__global__ void zero_u32_kernel(u32* __restrict__ p, int n) {
  int i = blockIdx.x * blockDim.x + threadIdx.x;
  if (i < n) p[i] = 0u;
}

__global__ void hist_kernel(const float* __restrict__ s, int n, u32* __restrict__ hist) {
  int i = blockIdx.x * blockDim.x + threadIdx.x;
  int stride = gridDim.x * blockDim.x;
  for (; i < n; i += stride) {
    u32 b = (__float_as_uint(s[i]) & 0x7FFFFFFFu) >> 16;
    atomicAdd(&hist[b], 1u);
  }
}

__global__ void select_kernel(const u32* __restrict__ hist, u32 j, u32* __restrict__ thr) {
  __shared__ u32 partial[256];
  __shared__ u32 segInfo[2];
  int t = threadIdx.x;
  u32 s = 0;
  for (int b = 0; b < 256; ++b) s += hist[t * 256 + b];
  partial[t] = s;
  __syncthreads();
  if (t == 0) {
    u32 cum = 0; u32 seg = 255; u32 before = 0;
    for (int i = 0; i < 256; ++i) {
      if (j < cum + partial[i]) { seg = (u32)i; before = cum; break; }
      cum += partial[i];
    }
    segInfo[0] = seg; segInfo[1] = before;
  }
  __syncthreads();
  if (t == 0) {
    u32 seg = segInfo[0], cum = segInfo[1];
    u32 T = seg * 256 + 255;
    for (int b = 0; b < 256; ++b) {
      u32 c = hist[seg * 256 + b];
      if (j < cum + c) { T = seg * 256 + (u32)b; break; }
      cum += c;
    }
    *thr = T;
  }
}

__global__ void pack_kernel(const float* __restrict__ w, const float* __restrict__ s,
                            const u32* __restrict__ thr, u16* __restrict__ out,
                            int M, int Ko, int Kp) {
  int idx = blockIdx.x * blockDim.x + threadIdx.x;
  if (idx >= M * Kp) return;
  int m = idx / Kp;
  int k = idx - m * Kp;
  u16 val = 0;
  if (k < Ko) {
    int src = m * Ko + k;
    u32 bin = (__float_as_uint(s[src]) & 0x7FFFFFFFu) >> 16;
    if (bin >= *thr) val = f2bf(w[src]);
  }
  out[idx] = val;
}

// ---------------- WMMA implicit-GEMM, double-buffered, async A staging ----------------
// Block: 256 threads = 8 wave32 (2 M x 4 N). Wave tile 64x64 -> block tile 128x256.
// LDS pitch 40 elements (80B) keeps async b128 stores 16B-aligned and staggers banks.
#define LDSP 40

template<typename InT, typename OutT, bool CONV_IN, bool CONV_OUT, bool RELU>
__global__ __launch_bounds__(256) void wmma_gemm(
    const u16* __restrict__ A, const InT* __restrict__ B,
    const float* __restrict__ bias, OutT* __restrict__ Cdst,
    int M, int N, int K,
    int Cin, int Hin, int Win, int stride, int woShift, int hwShift, int CoutCh)
{
  __shared__ u16 ldsA[2][128 * LDSP];
  __shared__ u16 ldsB[2][256 * LDSP];

  const int t     = threadIdx.x;
  const int lane  = t & 31;
  const int wid   = t >> 5;
  const int waveM = wid >> 2;      // 0..1
  const int waveN = wid & 3;       // 0..3
  const int hsel  = lane >> 4;     // lane half (ISA A/B layout)
  const int row   = lane & 15;

  const int blockM = blockIdx.y * 128;
  const int blockN = blockIdx.x * 256;

  const int HWout = 1 << hwShift;
  const int Wmask = (1 << woShift) - 1;

  // A loader coords: 2 threads per row, 2x16B async chunks each
  const int aLoc  = t >> 1;
  const int aRow  = blockM + aLoc;
  const int aOff  = (t & 1) * 16;
  // B loader coords: 1 thread per row, 32 elements
  const int bLoc  = t;
  const int bN    = blockN + bLoc;

  int bBatch = 0, bOy = 0, bOx = 0;
  if (CONV_IN) {
    int nn = (bN < N) ? bN : 0;
    bBatch = nn >> hwShift;
    int p = nn & (HWout - 1);
    bOy = p >> woShift;
    bOx = p & Wmask;
  }

  auto loadA = [&](int kt, int buf) {
    if (aRow < M) {
      const u16* g = A + (size_t)aRow * K + (kt << 5) + aOff;
      u16* l = &ldsA[buf][aLoc * LDSP + aOff];
      async_ld16(l, g);
      async_ld16(l + 8, g + 8);
    } else {
      u32* dst = (u32*)&ldsA[buf][aLoc * LDSP + aOff];
      #pragma unroll
      for (int i = 0; i < 8; ++i) dst[i] = 0u;
    }
  };

  auto loadB = [&](int kt, int buf) {
    if constexpr (CONV_IN) {
      u16* dst = &ldsB[buf][bLoc * LDSP];
      const int kBase = kt << 5;
      #pragma unroll 4
      for (int kk = 0; kk < 32; ++kk) {
        int k = kBase + kk;
        u16 val = 0;
        if (bN < N) {
          int cin = k / 9;
          int r9  = k - cin * 9;
          int ky  = r9 / 3;
          int kx  = r9 - ky * 3;
          int iy  = bOy * stride - 1 + ky;
          int ix  = bOx * stride - 1 + kx;
          if (cin < Cin && iy >= 0 && iy < Hin && ix >= 0 && ix < Win)
            val = toBF<InT>(B[(((size_t)bBatch * Cin + cin) * Hin + iy) * Win + ix]);
        }
        dst[kk] = val;
      }
    } else {
      if (bN < N) {
        const u16* g = (const u16*)B + (size_t)bN * K + (kt << 5);
        u16* l = &ldsB[buf][bLoc * LDSP];
        async_ld16(l, g);
        async_ld16(l + 8, g + 8);
        async_ld16(l + 16, g + 16);
        async_ld16(l + 24, g + 24);
      } else {
        u32* dst = (u32*)&ldsB[buf][bLoc * LDSP];
        #pragma unroll
        for (int i = 0; i < 16; ++i) dst[i] = 0u;
      }
    }
  };

  v8f acc[4][4] = {};

  const int ktiles = K >> 5;
  loadA(0, 0);
  loadB(0, 0);

  for (int kt = 0; kt < ktiles; ++kt) {
    const int cur = kt & 1;
    wait_asynccnt0();           // this wave's async stores to `cur` done
    __syncthreads();            // all waves' tile writes visible
    if (kt + 1 < ktiles) {      // prefetch next tile into the other buffer
      loadA(kt + 1, cur ^ 1);
      loadB(kt + 1, cur ^ 1);
    }

    union Frag { u32 u[8]; v16bf v; };
    Frag af[4];
    #pragma unroll
    for (int f = 0; f < 4; ++f) {
      const u32* src = (const u32*)&ldsA[cur][(waveM * 64 + f * 16 + row) * LDSP];
      #pragma unroll
      for (int v = 0; v < 8; ++v) {
        int kk = 2 * v + 16 * hsel;
        af[f].u[v] = src[kk >> 1];
      }
    }
    #pragma unroll
    for (int g = 0; g < 4; ++g) {
      Frag bfr;
      const u32* src = (const u32*)&ldsB[cur][(waveN * 64 + g * 16 + row) * LDSP];
      #pragma unroll
      for (int v = 0; v < 8; ++v) {
        int kk = 2 * v + 16 * hsel;
        bfr.u[v] = src[kk >> 1];
      }
      #pragma unroll
      for (int f = 0; f < 4; ++f)
        acc[f][g] = __builtin_amdgcn_wmma_f32_16x16x32_bf16(
            false, af[f].v, false, bfr.v, (short)0, acc[f][g], false, false);
    }
  }

  // ---- fused epilogue: bias + relu + layout-aware store ----
  #pragma unroll
  for (int f = 0; f < 4; ++f) {
    #pragma unroll
    for (int g = 0; g < 4; ++g) {
      #pragma unroll
      for (int i = 0; i < 8; ++i) {
        int m = blockM + waveM * 64 + f * 16 + i + 8 * hsel;   // C/D VGPR layout
        int n = blockN + waveN * 64 + g * 16 + row;
        if (m < M && n < N) {
          float v = acc[f][g][i] + bias[m];
          if (RELU) v = fmaxf(v, 0.0f);
          if constexpr (CONV_OUT) {
            int batch = n >> hwShift;
            int p = n & (HWout - 1);
            Cdst[((size_t)batch * CoutCh + m) * HWout + p] = fromF<OutT>(v);
          } else {
            Cdst[(size_t)n * M + m] = fromF<OutT>(v);
          }
        }
      }
    }
  }
}

// ---------------- global average pool: [64][512][16x16] bf16 -> [64][512] bf16 ----------------
__global__ void gap_kernel(const u16* __restrict__ h, u16* __restrict__ pooled) {
  int idx = blockIdx.x * blockDim.x + threadIdx.x;   // 0..32767 = n*512 + c
  const u16* p = h + (size_t)idx * 256;
  float s = 0.f;
  #pragma unroll 4
  for (int i = 0; i < 256; i += 4)
    s += bf2f(p[i]) + bf2f(p[i + 1]) + bf2f(p[i + 2]) + bf2f(p[i + 3]);
  pooled[idx] = f2bf(s * (1.0f / 256.0f));
}

static inline int imin_host(int a, int b) { return a < b ? a : b; }

extern "C" void kernel_launch(void* const* d_in, const int* in_sizes, int n_in,
                              void* d_out, int out_size, void* d_ws, size_t ws_size,
                              hipStream_t stream) {
  (void)in_sizes; (void)n_in; (void)out_size; (void)ws_size;
  const float* x   = (const float*)d_in[0];
  const float* w1  = (const float*)d_in[1];
  const float* s1  = (const float*)d_in[2];
  const float* b1  = (const float*)d_in[3];
  const float* w2  = (const float*)d_in[4];
  const float* s2  = (const float*)d_in[5];
  const float* b2  = (const float*)d_in[6];
  const float* w3  = (const float*)d_in[7];
  const float* s3  = (const float*)d_in[8];
  const float* b3  = (const float*)d_in[9];
  const float* fw1 = (const float*)d_in[10];
  const float* fs1 = (const float*)d_in[11];
  const float* fb1 = (const float*)d_in[12];
  const float* fw2 = (const float*)d_in[13];
  const float* fs2 = (const float*)d_in[14];
  const float* fb2 = (const float*)d_in[15];
  float* out = (float*)d_out;

  char* ws = (char*)d_ws;
  size_t o = 0;
  auto alloc = [&](size_t bytes) { size_t r = o; o = (o + bytes + 255) & ~(size_t)255; return r; };
  u32* hist = (u32*)(ws + alloc((size_t)65536 * 4));
  u32* thr  = (u32*)(ws + alloc(8 * 4));
  u16* w1p  = (u16*)(ws + alloc((size_t)128 * 32 * 2));
  u16* w2p  = (u16*)(ws + alloc((size_t)256 * 1152 * 2));
  u16* w3p  = (u16*)(ws + alloc((size_t)512 * 2304 * 2));
  u16* f1p  = (u16*)(ws + alloc((size_t)1024 * 512 * 2));
  u16* f2p  = (u16*)(ws + alloc((size_t)10 * 1024 * 2));
  u16* h1   = (u16*)(ws + alloc((size_t)64 * 128 * 64 * 64 * 2));
  u16* h2   = (u16*)(ws + alloc((size_t)64 * 256 * 32 * 32 * 2));
  u16* h3   = (u16*)(ws + alloc((size_t)64 * 512 * 16 * 16 * 2));
  u16* pool = (u16*)(ws + alloc((size_t)64 * 512 * 2));
  u16* fc1o = (u16*)(ws + alloc((size_t)64 * 1024 * 2));

  struct MaskJob { const float* s; const float* w; u16* dst; int M, Ko, Kp; u32* slot; };
  MaskJob jobs[5] = {
    { s1,  w1,  w1p, 128,  27,   32,   thr + 0 },
    { s2,  w2,  w2p, 256,  1152, 1152, thr + 1 },
    { s3,  w3,  w3p, 512,  2304, 2304, thr + 2 },
    { fs1, fw1, f1p, 1024, 512,  512,  thr + 3 },
    { fs2, fw2, f2p, 10,   1024, 1024, thr + 4 },
  };
  for (int i = 0; i < 5; ++i) {
    int n = jobs[i].M * jobs[i].Ko;
    u32 j = (u32)(n / 2);  // int((1-0.5)*n) zeroed
    zero_u32_kernel<<<256, 256, 0, stream>>>(hist, 65536);
    hist_kernel<<<imin_host((n + 255) / 256, 2048), 256, 0, stream>>>(jobs[i].s, n, hist);
    select_kernel<<<1, 256, 0, stream>>>(hist, j, jobs[i].slot);
    int np = jobs[i].M * jobs[i].Kp;
    pack_kernel<<<(np + 255) / 256, 256, 0, stream>>>(
        jobs[i].w, jobs[i].s, jobs[i].slot, jobs[i].dst, jobs[i].M, jobs[i].Ko, jobs[i].Kp);
  }

  // conv1: [64,3,64,64] -> [64,128,64,64], K padded 27->32
  {
    int M = 128, N = 64 * 64 * 64, K = 32;
    dim3 grid((N + 255) / 256, (M + 127) / 128);
    wmma_gemm<float, u16, true, true, true><<<grid, 256, 0, stream>>>(
        w1p, x, b1, h1, M, N, K, 3, 64, 64, 1, /*woShift*/6, /*hwShift*/12, 128);
  }
  // conv2: stride 2 -> [64,256,32,32], K = 128*9 = 1152
  {
    int M = 256, N = 64 * 32 * 32, K = 1152;
    dim3 grid((N + 255) / 256, (M + 127) / 128);
    wmma_gemm<u16, u16, true, true, true><<<grid, 256, 0, stream>>>(
        w2p, h1, b2, h2, M, N, K, 128, 64, 64, 2, /*woShift*/5, /*hwShift*/10, 256);
  }
  // conv3: stride 2 -> [64,512,16,16], K = 256*9 = 2304
  {
    int M = 512, N = 64 * 16 * 16, K = 2304;
    dim3 grid((N + 255) / 256, (M + 127) / 128);
    wmma_gemm<u16, u16, true, true, true><<<grid, 256, 0, stream>>>(
        w3p, h2, b3, h3, M, N, K, 256, 32, 32, 2, /*woShift*/4, /*hwShift*/8, 512);
  }
  // global average pool
  gap_kernel<<<128, 256, 0, stream>>>(h3, pool);
  // FC1: [64,512] @ [1024,512]^T -> [64,1024] (relu, bf16)
  {
    int M = 1024, N = 64, K = 512;
    dim3 grid((N + 255) / 256, (M + 127) / 128);
    wmma_gemm<u16, u16, false, false, true><<<grid, 256, 0, stream>>>(
        f1p, pool, fb1, fc1o, M, N, K, 0, 0, 0, 0, 0, 0, 0);
  }
  // FC2: [64,1024] @ [10,1024]^T -> [64,10] fp32 out
  {
    int M = 10, N = 64, K = 1024;
    dim3 grid((N + 255) / 256, (M + 127) / 128);
    wmma_gemm<u16, float, false, false, false><<<grid, 256, 0, stream>>>(
        f2p, fc1o, fb2, out, M, N, K, 0, 0, 0, 0, 0, 0, 0);
  }
}
